// DistanceScaledDotAttention_15607911154068
// MI455X (gfx1250) — compile-verified, matrix operand order checked by reading, not run on hardware
//
#include <hip/hip_runtime.h>
#include <hip/hip_bf16.h>

// DistanceScaledDotAttention for MI455X (gfx1250, wave32, WMMA).
// B=4, H=8, S=2048, D=64.
// d_in: Q f32[B,H,S,D], K f32[B,H,S,D], V f32[B,H,S,D],
//       dist_factor f32[B,H,S,S], attn_mask bool(1B)[B,H,S,S]
// d_out: context f32[B,H,S,D] ++ attn f32[B,H,S,S]

typedef _Float16 v16h __attribute__((ext_vector_type(16)));
typedef float    v8f  __attribute__((ext_vector_type(8)));

#define S_DIM 2048
#define D_DIM 64
#define SPAD  2052            // LDS row stride; 2052 % 64 == 4 -> bank-conflict-free softmax
#define NEG_INF_VAL (-1e10f)

// 16-bit A/B fragment K-index for element e (0..15), lane-group g (=lane/16):
// VGPR j<4 holds K {0..7|8..15}, VGPR j>=4 holds K {16..23|24..31}  (ISA 7.12.2)
__device__ __forceinline__ int frag_k(int e, int g) {
    return ((e >> 3) << 4) + (g << 3) + (e & 7);
}

// Issue all global loads for one k-tile (K fragment rows + dist/mask epilogue data).
#define LOAD_TILE(KT, KF, DF, MK)                                              \
    do {                                                                       \
        const float* Krow_ = Kb + (size_t)((((KT) << 4)) + ln) * D_DIM;        \
        _Pragma("unroll")                                                      \
        for (int e_ = 0; e_ < 16; ++e_) {                                      \
            const int d_ = frag_k(e_, g);                                      \
            (KF)[e_]      = Krow_[d_];                                         \
            (KF)[e_ + 16] = Krow_[d_ + 32];                                    \
        }                                                                      \
        const int gcol_ = ((KT) << 4) + ln;                                    \
        _Pragma("unroll")                                                      \
        for (int v_ = 0; v_ < 8; ++v_) {                                       \
            const size_t off_ = (size_t)(v_ + (g << 3)) * S_DIM + gcol_;       \
            (DF)[v_] = dist_wg[off_];                                          \
            (MK)[v_] = mask_wg[off_];                                          \
        }                                                                      \
    } while (0)

__global__ __launch_bounds__(256)
void dsda_kernel(const float* __restrict__ Q,
                 const float* __restrict__ K,
                 const float* __restrict__ V,
                 const float* __restrict__ dist,
                 const unsigned char* __restrict__ mask,
                 float* __restrict__ ctx_out,
                 float* __restrict__ attn_out) {
    extern __shared__ float smem[];
    float* sc     = smem;                  // [16][SPAD] score/exp strip (131328 B)
    float* cpart  = smem + 16 * SPAD;      // [4][4][16][16] partial context (16384 B)
    float* red    = cpart + 4096;          // [16][16] reduction scratch
    float* rowmax = red + 256;             // [16]
    float* rowinv = rowmax + 16;           // [16]

    const int tid  = threadIdx.x;
    const int lane = tid & 31;
    const int wv   = tid >> 5;     // wave id 0..7
    const int g    = lane >> 4;    // lane group 0/1
    const int ln   = lane & 15;    // row/col within fragment

    const int bh = blockIdx.x >> 7;        // 0..31  (B*H)
    const int qt = blockIdx.x & 127;       // q tile 0..127
    const int q0 = qt << 4;                // first q row of strip

    const size_t sd_base = (size_t)bh * S_DIM * D_DIM;
    const size_t ss_base = ((size_t)bh * S_DIM + (size_t)q0) * S_DIM;

    const float*         dist_wg = dist + ss_base;
    const unsigned char* mask_wg = mask + ss_base;
    const float*         Kb      = K + sd_base;

    // ---------------- Phase A: scores = mask( dist * (Q K^T)/8 ) ------------
    // A fragments for Q rows q0..q0+15 (K-dim = D = 64 -> two v16h)
    v16h aq0, aq1;
    {
        const float* Qr = Q + sd_base + (size_t)(q0 + ln) * D_DIM;
#pragma unroll
        for (int e = 0; e < 16; ++e) {
            const int d = frag_k(e, g);
            aq0[e] = (_Float16)Qr[d];
            aq1[e] = (_Float16)Qr[d + 32];
        }
    }

    // 1-deep software pipeline over this wave's 16 k-tiles (kt = wv + 8*it):
    // loads for tile it+1 are in flight while tile it runs WMMA + epilogue.
    {
        float kf0[32], df0[8];
        unsigned char mk0[8];
        LOAD_TILE(wv, kf0, df0, mk0);

#pragma unroll 1
        for (int it = 0; it < 16; ++it) {
            const int kt = wv + (it << 3);
            float kf1[32], df1[8];
            unsigned char mk1[8];
            if (it < 15) LOAD_TILE(kt + 8, kf1, df1, mk1);

            // convert current K tile to f16 fragments
            v16h bk0, bk1;
#pragma unroll
            for (int e = 0; e < 16; ++e) {
                bk0[e] = (_Float16)kf0[e];
                bk1[e] = (_Float16)kf0[e + 16];
            }
            v8f c = {};
            c = __builtin_amdgcn_wmma_f32_16x16x32_f16(false, aq0, false, bk0,
                                                       (short)0, c, false, false);
            c = __builtin_amdgcn_wmma_f32_16x16x32_f16(false, aq1, false, bk1,
                                                       (short)0, c, false, false);
            // epilogue: scale, distance factor, mask; park in padded LDS strip
            const int gcol = (kt << 4) + ln;
#pragma unroll
            for (int v = 0; v < 8; ++v) {
                const int m = v + (g << 3);
                float s = c[v] * 0.125f * df0[v];      // 1/sqrt(64) = 0.125
                if (mk0[v]) s = NEG_INF_VAL;
                sc[m * SPAD + gcol] = s;
            }
            if (it < 15) {
#pragma unroll
                for (int e = 0; e < 32; ++e) kf0[e] = kf1[e];
#pragma unroll
                for (int v = 0; v < 8; ++v) { df0[v] = df1[v]; mk0[v] = mk1[v]; }
            }
        }
    }
    __syncthreads();

    // ---------------- Phase B: row softmax over the LDS strip ---------------
    const int r  = tid & 15;      // row
    const int cg = tid >> 4;      // column group (0..15), columns cg + 16*i
    {
        float mx = -3.402823e38f;
#pragma unroll 4
        for (int i = 0; i < 128; ++i)
            mx = fmaxf(mx, sc[r * SPAD + cg + (i << 4)]);
        red[r * 16 + cg] = mx;
    }
    __syncthreads();
    if (tid < 16) {
        float m2 = red[tid * 16];
#pragma unroll
        for (int j = 1; j < 16; ++j) m2 = fmaxf(m2, red[tid * 16 + j]);
        rowmax[tid] = m2;
    }
    __syncthreads();
    {
        const float rm = rowmax[r];
        float sum = 0.f;
#pragma unroll 4
        for (int i = 0; i < 128; ++i) {
            const int idx = r * SPAD + cg + (i << 4);
            const float e = __expf(sc[idx] - rm);
            sc[idx] = e;                      // keep UNNORMALIZED exp in LDS
            sum += e;
        }
        red[r * 16 + cg] = sum;
    }
    __syncthreads();
    if (tid < 16) {
        float s2 = 0.f;
#pragma unroll
        for (int j = 0; j < 16; ++j) s2 += red[tid * 16 + j];
        rowinv[tid] = 1.0f / s2;
    }
    __syncthreads();

    // single coalesced normalized attn store (strip rows contiguous in global)
    {
        float* attn_wg = attn_out + ss_base;
        for (int idx = tid; idx < 16 * S_DIM; idx += 256) {
            const int rr = idx >> 11;
            attn_wg[idx] = sc[rr * SPAD + (idx & (S_DIM - 1))] * rowinv[rr];
        }
    }

    // ---------------- Phase C: context = attn @ V ---------------------------
    // 8 waves = 4 K-quarters (512 each) x 2 N-pairs; each wave shares one
    // A fragment across two independent WMMA accumulator chains.
    {
        const int kh = wv >> 1;            // K quarter 0..3
        const int np = wv & 1;             // N pair: cols [np*32, np*32+32)
        const int dcol = (np << 5) + ln;
        const float* Vb = V + sd_base;
        const float rinv = rowinv[ln];     // normalization folded into A
        v8f acc0 = {}, acc1 = {};
        for (int kk = 0; kk < 16; ++kk) {
            const int kbase = (kh << 9) + (kk << 5);
            v16h a, b0, b1;
#pragma unroll
            for (int e = 0; e < 16; ++e) {
                const int n = kbase + frag_k(e, g);
                b0[e] = (_Float16)Vb[(size_t)n * D_DIM + dcol];
                b1[e] = (_Float16)Vb[(size_t)n * D_DIM + dcol + 16];
                a[e]  = (_Float16)(sc[ln * SPAD + n] * rinv);
            }
            acc0 = __builtin_amdgcn_wmma_f32_16x16x32_f16(false, a, false, b0,
                                                          (short)0, acc0, false, false);
            acc1 = __builtin_amdgcn_wmma_f32_16x16x32_f16(false, a, false, b1,
                                                          (short)0, acc1, false, false);
        }
        const int nt0 = np << 1, nt1 = nt0 + 1;
#pragma unroll
        for (int v = 0; v < 8; ++v) {
            const int m = v + (g << 3);
            cpart[(kh << 10) + (nt0 << 8) + (m << 4) + ln] = acc0[v];
            cpart[(kh << 10) + (nt1 << 8) + (m << 4) + ln] = acc1[v];
        }
    }
    __syncthreads();

    // reduce the four K-quarters, coalesced context store
    {
        float* ctx_wg = ctx_out + sd_base + (size_t)q0 * D_DIM;
        for (int idx = tid; idx < 16 * D_DIM; idx += 256) {
            const int m  = idx >> 6;
            const int d  = idx & 63;
            const int nt = d >> 4;
            const int cc = d & 15;
            const int base = (nt << 8) + (m << 4) + cc;
            ctx_wg[idx] = cpart[base] + cpart[1024 + base] +
                          cpart[2048 + base] + cpart[3072 + base];
        }
    }
}

extern "C" void kernel_launch(void* const* d_in, const int* in_sizes, int n_in,
                              void* d_out, int out_size, void* d_ws, size_t ws_size,
                              hipStream_t stream) {
    (void)in_sizes; (void)n_in; (void)d_ws; (void)ws_size; (void)out_size;
    constexpr int B = 4, H = 8;
    const float*         Q    = (const float*)d_in[0];
    const float*         K    = (const float*)d_in[1];
    const float*         V    = (const float*)d_in[2];
    const float*         dist = (const float*)d_in[3];
    const unsigned char* mask = (const unsigned char*)d_in[4];   // numpy bool = 1 byte

    float* ctx  = (float*)d_out;
    float* attn = (float*)d_out + (size_t)B * H * S_DIM * D_DIM;

    const int blocks = B * H * (S_DIM / 16);                     // 4096
    const size_t smem = (size_t)(16 * SPAD + 4096 + 256 + 32) * sizeof(float); // ~145 KB
    dsda_kernel<<<blocks, 256, smem, stream>>>(Q, K, V, dist, mask, ctx, attn);
}
